// BaseModel_59322088292901
// MI455X (gfx1250) — matrix operand order repaired and checked by reading, not compiled, not run on hardware
//
#include <hip/hip_runtime.h>
#include <cstdint>
#include <cstddef>

typedef unsigned short u16;
typedef __attribute__((ext_vector_type(16))) __bf16 v16bf;
typedef __attribute__((ext_vector_type(8)))  float  v8f;
typedef int v4i_vs __attribute__((__vector_size__(16)));   // matches builtin param pointee

#define BB 2048
#define LL 100
#define FF 64
#define EE 256
#define DD 256
#define FCN 128

// ---------------- bf16 helpers ----------------
__device__ __forceinline__ u16 f2bf(float x) {
    union { float f; uint32_t u; } v; v.f = x;
    uint32_t r = v.u + 0x7FFFu + ((v.u >> 16) & 1u);   // round-to-nearest-even
    return (u16)(r >> 16);
}
__device__ __forceinline__ float bf2f(u16 h) {
    union { uint32_t u; float f; } v; v.u = ((uint32_t)h) << 16;
    return v.f;
}
__device__ __forceinline__ float sigmoidf(float x) { return 1.f / (1.f + __expf(-x)); }

// ---------------- async global->LDS copy (CDNA5 ASYNCcnt path) ----------------
#if __has_builtin(__builtin_amdgcn_global_load_async_to_lds_b128)
#define HAS_ASYNC_LDS 1
#else
#define HAS_ASYNC_LDS 0
#endif

__device__ __forceinline__ void cp_b128(const u16* gsrc, u16* ldst) {
#if HAS_ASYNC_LDS
    __builtin_amdgcn_global_load_async_to_lds_b128(
        (__attribute__((address_space(1))) v4i_vs*)(gsrc),
        (__attribute__((address_space(3))) v4i_vs*)(ldst), 0, 0);
#else
    *(uint4*)ldst = *(const uint4*)gsrc;
#endif
}
__device__ __forceinline__ void cp_wait() {
#if HAS_ASYNC_LDS
#if __has_builtin(__builtin_amdgcn_s_wait_asynccnt)
    __builtin_amdgcn_s_wait_asynccnt(0);
#else
    asm volatile("s_wait_asynccnt 0x0" ::: "memory");
#endif
#endif
}

union Frag16 { v16bf v; uint4 u[2]; };

// A fragment (16x32 bf16, MxK) per ISA layout:
// lanes 0-15: M=lane, K in {0..7} (u[0]) and {16..23} (u[1])
// lanes 16-31: M=lane-16, K in {8..15} and {24..31}
__device__ __forceinline__ v16bf load_a_frag(const u16* As /*[rows][32]*/, int row0, int lane) {
    int r  = row0 + (lane & 15);
    int kb = (lane >> 4) * 8;
    Frag16 f;
    f.u[0] = *(const uint4*)(As + r * 32 + kb);
    f.u[1] = *(const uint4*)(As + r * 32 + kb + 16);
    return f.v;
}
// B fragment (32x16 bf16, KxN) from LDS staged as Bt[n][k] (weights pre-transposed):
// lanes 0-15: N=lane, K in {0..7} and {16..23} ; lanes 16-31: N=lane-16, K in {8..15} and {24..31}
__device__ __forceinline__ v16bf load_b_frag(const u16* Bt /*[ncols][32]*/, int ncol0, int lane) {
    int n  = ncol0 + (lane & 15);
    int ks = (lane >> 4) * 8;
    Frag16 f;
    f.u[0] = *(const uint4*)(Bt + n * 32 + ks);
    f.u[1] = *(const uint4*)(Bt + n * 32 + ks + 16);
    return f.v;
}

// ---------------- generic 64x64-tile WMMA GEMM, dual-A (concat along K) ---------
// C[M,N] = act( [A0 | A1] @ BT^T + bias ), BT is N x K (pre-transposed weights).
// A0 rows supply k in [0,K0), A1 rows supply k in [K0,K). grid=(N/64, M/64), 256 thr.
__global__ __launch_bounds__(256)
void gemm_wmma_bf16(const u16* __restrict__ A0, int lda0, int K0,
                    const u16* A1, int lda1,
                    const u16* __restrict__ BT, int ldk,
                    const float* __restrict__ bias,
                    float* __restrict__ C, int ldc,
                    int K, int act)
{
    __shared__ __align__(16) u16 As[64 * 32];
    __shared__ __align__(16) u16 Bs[64 * 32];
    int tid  = threadIdx.x;
    int m0   = blockIdx.y * 64, n0 = blockIdx.x * 64;
    int lane = tid & 31, w = tid >> 5;
    int mt = w & 3, nt0 = (w >> 2) * 2;
    int r = tid >> 2, c = (tid & 3) * 8;

    v8f acc[2] = {};
    for (int kb = 0; kb < K; kb += 32) {
        // stage A tile (64 x 32): pick source operand by k (wave-uniform)
        const u16* Ap; int lda_, kloc;
        if (kb < K0) { Ap = A0; lda_ = lda0; kloc = kb; }
        else         { Ap = A1; lda_ = lda1; kloc = kb - K0; }
        cp_b128(&Ap[(size_t)(m0 + r) * lda_ + kloc + c], &As[r * 32 + c]);
        // stage B tile (64 n-rows x 32 k): contiguous, already transposed
        cp_b128(&BT[(size_t)(n0 + r) * ldk + kb + c], &Bs[r * 32 + c]);
        cp_wait();
        __syncthreads();
        v16bf af = load_a_frag(As, mt * 16, lane);
#pragma unroll
        for (int j = 0; j < 2; j++) {
            v16bf bfv = load_b_frag(Bs, (nt0 + j) * 16, lane);
            acc[j] = __builtin_amdgcn_wmma_f32_16x16x32_bf16(false, af, false, bfv,
                                                             (short)0, acc[j], false, false);
        }
        __syncthreads();
    }
    int rowoff = (lane >> 4) * 8, col15 = lane & 15;
#pragma unroll
    for (int j = 0; j < 2; j++) {
        int ncol = n0 + (nt0 + j) * 16 + col15;
        float bv = bias ? bias[ncol] : 0.f;
#pragma unroll
        for (int v = 0; v < 8; v++) {
            int row = m0 + mt * 16 + rowoff + v;
            float xv = acc[j][v] + bv;
            if (act) xv = tanhf(xv);
            C[(size_t)row * ldc + ncol] = xv;
        }
    }
}

// ---------------- decoder temporal-attention scores ----------------
// scores[r] = Wl . tanh( enc_bf[r,:] @ WdTop + hWd[r % B, :] ) + bl , r = t*B+b
// WdTt is the full Wd transposed: [256 n][512 k]; top half is k in [0,256).
__global__ __launch_bounds__(256)
void dec_scores_kernel(const u16* __restrict__ encbf,  // (L*B) x 256 bf16
                       const u16* __restrict__ WdTt,   // 256 x 512 bf16 (N x K)
                       const float* __restrict__ hWd,  // B x 256 (includes bd)
                       const float* __restrict__ Wl,   // 256
                       const float* __restrict__ bl,   // 1
                       float* __restrict__ scores)     // L*B
{
    __shared__ __align__(16) u16 As[64 * 32];
    __shared__ __align__(16) u16 Bs[256 * 32];
    __shared__ float scp[2][64];
    int tid  = threadIdx.x;
    int m0   = blockIdx.x * 64;
    int lane = tid & 31, w = tid >> 5;
    int mt = w & 3, ntBase = (w >> 2) * 8, grp = w >> 2;
    int r = tid >> 2, c = (tid & 3) * 8;

    v8f acc[8] = {};
    for (int kb = 0; kb < 256; kb += 32) {
        cp_b128(&encbf[(size_t)(m0 + r) * 256 + kb + c], &As[r * 32 + c]);
#pragma unroll
        for (int cc = 0; cc < 4; cc++) {
            int n = r + cc * 64;
            cp_b128(&WdTt[(size_t)n * 512 + kb + c], &Bs[n * 32 + c]);
        }
        cp_wait();
        __syncthreads();
        v16bf af = load_a_frag(As, mt * 16, lane);
#pragma unroll
        for (int j = 0; j < 8; j++) {
            v16bf bfv = load_b_frag(Bs, (ntBase + j) * 16, lane);
            acc[j] = __builtin_amdgcn_wmma_f32_16x16x32_bf16(false, af, false, bfv,
                                                             (short)0, acc[j], false, false);
        }
        __syncthreads();
    }
    int rowoff = (lane >> 4) * 8, col15 = lane & 15;
    float psum[8] = {};
#pragma unroll
    for (int j = 0; j < 8; j++) {
        int ncol = (ntBase + j) * 16 + col15;
        float wl = Wl[ncol];
#pragma unroll
        for (int v = 0; v < 8; v++) {
            int row = m0 + mt * 16 + rowoff + v;
            int b   = row & (BB - 1);
            float xv = tanhf(acc[j][v] + hWd[(size_t)b * 256 + ncol]);
            psum[v] += xv * wl;
        }
    }
#pragma unroll
    for (int v = 0; v < 8; v++) {   // reduce across the 16 lanes sharing a row
        float pr = psum[v];
        pr += __shfl_xor(pr, 1, 32);
        pr += __shfl_xor(pr, 2, 32);
        pr += __shfl_xor(pr, 4, 32);
        pr += __shfl_xor(pr, 8, 32);
        if (col15 == 0) scp[grp][mt * 16 + rowoff + v] = pr;   // unique slots: deterministic
    }
    __syncthreads();
    if (tid < 64) scores[m0 + tid] = scp[0][tid] + scp[1][tid] + bl[0];
}

// ---------------- input attention: softmax(energies) * x_t ----------------
__global__ __launch_bounds__(256)
void attn_softmax_kernel(const float* __restrict__ energies, // B x 64 (post-tanh)
                         const float* __restrict__ x,        // B x L x 64
                         u16* __restrict__ encin,            // B x 64 bf16
                         int t)
{
    int lane = threadIdx.x & 31, w = threadIdx.x >> 5;
    int b = blockIdx.x * 8 + w;
    float e0 = energies[b * 64 + lane];
    float e1 = energies[b * 64 + lane + 32];
    float m = fmaxf(e0, e1);
    for (int off = 16; off > 0; off >>= 1) m = fmaxf(m, __shfl_xor(m, off, 32));
    float a0 = __expf(e0 - m), a1 = __expf(e1 - m);
    float s = a0 + a1;
    for (int off = 16; off > 0; off >>= 1) s += __shfl_xor(s, off, 32);
    float inv = 1.f / s;
    const float* xp = x + ((size_t)b * LL + t) * 64;
    encin[b * 64 + lane]      = f2bf(a0 * inv * xp[lane]);
    encin[b * 64 + lane + 32] = f2bf(a1 * inv * xp[lane + 32]);
}

// ---------------- temporal softmax over L + weighted context ----------------
__global__ __launch_bounds__(256)
void softmax_ctx_kernel(const float* __restrict__ scores, // L x B
                        const u16* __restrict__ encbf,    // (L*B) x 256 bf16
                        u16* __restrict__ ctxbf)          // B x 256 bf16
{
    __shared__ float alpha[8][LL];
    int lane = threadIdx.x & 31, w = threadIdx.x >> 5;
    int b = blockIdx.x * 8 + w;
    float m = -1e30f;
    for (int t = lane; t < LL; t += 32) m = fmaxf(m, scores[t * BB + b]);
    for (int off = 16; off > 0; off >>= 1) m = fmaxf(m, __shfl_xor(m, off, 32));
    float s = 0.f;
    for (int t = lane; t < LL; t += 32) {
        float e = __expf(scores[t * BB + b] - m);
        alpha[w][t] = e; s += e;
    }
    __syncthreads();
    for (int off = 16; off > 0; off >>= 1) s += __shfl_xor(s, off, 32);
    float inv = 1.f / s;
    int e0 = lane * 8;
    float acc[8] = {};
    for (int t = 0; t < LL; t++) {
        float wt = alpha[w][t] * inv;
        uint4 d = *(const uint4*)&encbf[((size_t)t * BB + b) * 256 + e0];
        const u16* ds = (const u16*)&d;
#pragma unroll
        for (int e = 0; e < 8; e++) acc[e] += wt * bf2f(ds[e]);
    }
    u16 ov[8];
#pragma unroll
    for (int e = 0; e < 8; e++) ov[e] = f2bf(acc[e]);
    *(uint4*)&ctxbf[(size_t)b * 256 + e0] = *(const uint4*)ov;
}

// ---------------- LSTM cell elementwise update ----------------
__global__ __launch_bounds__(256)
void lstm_update_kernel(const float* __restrict__ gates, // B x 1024 (i,f,g,o)
                        float* __restrict__ c,           // B x 256
                        u16* __restrict__ hbf,           // B x 256 bf16
                        u16* __restrict__ encslice)      // B x 256 bf16 or null
{
    int idx = blockIdx.x * 256 + threadIdx.x;   // b*256+e
    int b = idx >> 8, e = idx & 255;
    const float* g = gates + (size_t)b * 1024;
    float gi = g[e], gf = g[256 + e], gg = g[512 + e], go = g[768 + e];
    float cn = sigmoidf(gf) * c[idx] + sigmoidf(gi) * tanhf(gg);
    float hn = sigmoidf(go) * tanhf(cn);
    c[idx] = cn;
    u16 hb = f2bf(hn);
    hbf[idx] = hb;
    if (encslice) encslice[idx] = hb;
}

// ---------------- output head: tanh(fc @ Wo + bo) ----------------
__global__ __launch_bounds__(256)
void head_out_kernel(const float* __restrict__ fc, const float* __restrict__ Wo,
                     const float* __restrict__ bo, float* __restrict__ out, int j)
{
    int lane = threadIdx.x & 31, w = threadIdx.x >> 5;
    int b = blockIdx.x * 8 + w;
    const float* f = fc + (size_t)b * FCN;
    float s = f[lane] * Wo[lane] + f[lane + 32] * Wo[lane + 32] +
              f[lane + 64] * Wo[lane + 64] + f[lane + 96] * Wo[lane + 96];
    for (int off = 16; off > 0; off >>= 1) s += __shfl_xor(s, off, 32);
    if (lane == 0) out[b * 3 + j] = tanhf(s + bo[0]);
}

// ---------------- utility kernels ----------------
__global__ void f2bf_kernel(const float* __restrict__ src, u16* __restrict__ dst, int n) {
    int i = blockIdx.x * 256 + threadIdx.x;
    if (i < n) dst[i] = f2bf(src[i]);
}
// convert + transpose: src is K x N row-major f32; dst[n*ldK + kOff + k] = bf16(src[k,n])
__global__ void f2bf_t_kernel(const float* __restrict__ src, u16* __restrict__ dst,
                              int K, int N, int ldK, int kOff) {
    int i = blockIdx.x * 256 + threadIdx.x;
    if (i < K * N) {
        int k = i / N, n = i - k * N;
        dst[(size_t)n * ldK + kOff + k] = f2bf(src[i]);
    }
}
__global__ void fill_zero_kernel(uint32_t* __restrict__ p, int n) {
    int i = blockIdx.x * 256 + threadIdx.x;
    if (i < n) p[i] = 0u;
}

// ---------------- host ----------------
extern "C" void kernel_launch(void* const* d_in, const int* in_sizes, int n_in,
                              void* d_out, int out_size, void* d_ws, size_t ws_size,
                              hipStream_t stream)
{
    (void)in_sizes; (void)n_in; (void)out_size; (void)ws_size;
    const float* x    = (const float*)d_in[0];
    const float* Wa   = (const float*)d_in[1];
    const float* ba   = (const float*)d_in[2];
    const float* Wi   = (const float*)d_in[3];
    const float* Wh   = (const float*)d_in[4];
    const float* be   = (const float*)d_in[5];
    const float* Wd   = (const float*)d_in[6];
    const float* bd   = (const float*)d_in[7];
    const float* Wl   = (const float*)d_in[8];
    const float* bl   = (const float*)d_in[9];
    const float* Wdi  = (const float*)d_in[10];
    const float* Wdh  = (const float*)d_in[11];
    const float* bdec = (const float*)d_in[12];
    const float* Wf   = (const float*)d_in[13];
    const float* bf   = (const float*)d_in[14];
    const float* Wo   = (const float*)d_in[15];
    const float* bo   = (const float*)d_in[16];
    float* out = (float*)d_out;

    char* p = (char*)d_ws;
    auto take = [&](size_t bytes) { char* r = p; p += (bytes + 255) & ~(size_t)255; return r; };

    u16*   xbf    = (u16*)take((size_t)BB * LL * FF * 2);
    u16*   WaT    = (u16*)take((size_t)FF * 320 * 2);        // [64 n][320 k]
    u16*   WiWhT  = (u16*)take((size_t)1024 * 320 * 2);      // [1024 n][320 k] = [Wi;Wh]^T
    u16*   WdTt   = (u16*)take((size_t)DD * 512 * 2);        // [256 n][512 k] = Wd^T
    u16*   WdiDhT = (u16*)take((size_t)1024 * 512 * 2);      // [1024 n][512 k] = [Wdi;Wdh]^T
    u16*   WfT    = (u16*)take((size_t)FCN * DD * 2);        // [128 n][256 k]
    u16*   encout = (u16*)take((size_t)LL * BB * EE * 2);
    u16*   hbf    = (u16*)take((size_t)BB * EE * 2);
    float* cenc   = (float*)take((size_t)BB * EE * 4);
    float* energ  = (float*)take((size_t)BB * FF * 4);
    u16*   encin  = (u16*)take((size_t)BB * FF * 2);
    float* gates  = (float*)take((size_t)BB * 4 * EE * 4);
    float* hWd    = (float*)take((size_t)BB * DD * 4);
    float* scores = (float*)take((size_t)LL * BB * 4);
    u16*   ctxbf  = (u16*)take((size_t)BB * EE * 2);
    u16*   hdbf   = (u16*)take((size_t)BB * DD * 2);
    float* cdec   = (float*)take((size_t)BB * DD * 4);
    float* fc     = (float*)take((size_t)BB * FCN * 4);

    auto convT = [&](const float* s, u16* d, int K, int N, int ldK, int kOff) {
        f2bf_t_kernel<<<(K * N + 255) / 256, 256, 0, stream>>>(s, d, K, N, ldK, kOff);
    };
    auto zero = [&](void* d, size_t bytes) {
        int n = (int)(bytes / 4);
        fill_zero_kernel<<<(n + 255) / 256, 256, 0, stream>>>((uint32_t*)d, n);
    };
    auto gemm = [&](const u16* A0, int lda0, int K0, const u16* A1, int lda1,
                    const u16* BT, int ldk, const float* bias,
                    float* C, int ldc, int M, int N, int K, int act) {
        gemm_wmma_bf16<<<dim3(N / 64, M / 64), 256, 0, stream>>>(A0, lda0, K0, A1, lda1,
                                                                 BT, ldk, bias, C, ldc, K, act);
    };

    // ---- weight conversion to bf16, pre-transposed to N x K ----
    f2bf_kernel<<<(BB * LL * FF + 255) / 256, 256, 0, stream>>>(x, xbf, BB * LL * FF);
    convT(Wa,  WaT,    320, 64,   320, 0);
    convT(Wi,  WiWhT,  64,  1024, 320, 0);
    convT(Wh,  WiWhT,  256, 1024, 320, 64);
    convT(Wd,  WdTt,   512, 256,  512, 0);
    convT(Wdi, WdiDhT, 256, 1024, 512, 0);
    convT(Wdh, WdiDhT, 256, 1024, 512, 256);
    convT(Wf,  WfT,    256, 128,  256, 0);

    // ---- zero initial states (bf16 zero bits == 0.0) ----
    zero(hbf,  (size_t)BB * EE * 2);
    zero(cenc, (size_t)BB * EE * 4);
    zero(hdbf, (size_t)BB * DD * 2);
    zero(cdec, (size_t)BB * DD * 4);

    // ---- encoder: 100 sequential steps ----
    for (int t = 0; t < LL; t++) {
        // energies = tanh([x_t | h] @ Wa + ba): one K=320 concat GEMM
        gemm(xbf + (size_t)t * FF, LL * FF, FF, hbf, EE,
             WaT, 320, ba, energ, FF, BB, FF, 320, 1);
        attn_softmax_kernel<<<BB / 8, 256, 0, stream>>>(energ, x, encin, t);
        // gates = [enc_in | h] @ [Wi;Wh] + be: one K=320 concat GEMM
        gemm(encin, FF, FF, hbf, EE,
             WiWhT, 320, be, gates, 4 * EE, BB, 4 * EE, 320, 0);
        lstm_update_kernel<<<BB, 256, 0, stream>>>(gates, cenc, hbf,
                                                   encout + (size_t)t * BB * EE);
    }

    // ---- decoder: 3 output steps ----
    for (int j = 0; j < 3; j++) {
        // hWd = h_d @ Wd[bottom] + bd  (k offset 256 in WdTt)
        gemm(hdbf, DD, DD, hdbf, DD,
             WdTt + 256, 512, bd, hWd, DD, BB, DD, DD, 0);
        // scores[t,b] = Wl . tanh(enc_out @ Wd[top] + hWd) + bl
        dec_scores_kernel<<<(LL * BB) / 64, 256, 0, stream>>>(encout, WdTt, hWd,
                                                              Wl, bl, scores);
        // alpha = softmax over t ; ctx = sum alpha * enc_out
        softmax_ctx_kernel<<<BB / 8, 256, 0, stream>>>(scores, encout, ctxbf);
        // gates = [ctx | h_d] @ [Wdi;Wdh] + bdec: one K=512 concat GEMM
        gemm(ctxbf, EE, EE, hdbf, DD,
             WdiDhT, 512, bdec, gates, 4 * DD, BB, 4 * DD, 512, 0);
        lstm_update_kernel<<<BB, 256, 0, stream>>>(gates, cdec, hdbf, nullptr);
        // fc = tanh(h_d @ Wf + bf) ; out = tanh(fc @ Wo + bo)
        gemm(hdbf, DD, DD, hdbf, DD,
             WfT, 256, bf, fc, FCN, BB, FCN, 256, 1);
        head_out_kernel<<<BB / 8, 256, 0, stream>>>(fc, Wo, bo, out, j);
    }
}
